// BiLSTM_CRF_55722905699236
// MI455X (gfx1250) — compile-verified, matrix operand order checked by reading, not run on hardware
//
#include <hip/hip_runtime.h>
#include <hip/hip_bf16.h>

// Problem constants
#define TB 512
#define NB 64
#define HD 256
#define ED 256
#define G4 1024
#define KK 9
#define BT (NB * TB)

typedef __attribute__((ext_vector_type(16))) __bf16 v16bf;
typedef __attribute__((ext_vector_type(8)))  float  v8f;

struct B32x8 { uint4 lo; uint4 hi; };

__device__ __forceinline__ unsigned short f2bf(float f) {
  unsigned int u = __builtin_bit_cast(unsigned int, f);
  u += 0x7FFFu + ((u >> 16) & 1u);           // round-to-nearest-even
  return (unsigned short)(u >> 16);
}

// A/B fragment loader for v_wmma_f32_16x16x32_bf16 (wave32).
// lane L holds row (L&15); elements 0..7 are K = k0 + (L>>4)*8 + 0..7,
// elements 8..15 are K = k0 + 16 + (L>>4)*8 + 0..7 -> two 16B chunks/lane.
// B mirrors A with N=(L&15) indexing rows of the row-major weight matrix.
__device__ __forceinline__ v16bf load_frag(const __bf16* base, int stride, int lane, int k0) {
  const int half = lane >> 4;
  const __bf16* p = base + (size_t)(lane & 15) * stride + k0 + half * 8;
  B32x8 d;
  d.lo = *(const uint4*)(p);
  d.hi = *(const uint4*)(p + 16);
  return __builtin_bit_cast(v16bf, d);
}

__device__ __forceinline__ v8f wmma_bf16(v16bf a, v16bf b, v8f c) {
  return __builtin_amdgcn_wmma_f32_16x16x32_bf16(false, a, false, b, (short)0, c, false, false);
}
__device__ __forceinline__ float sigmoidf_(float x) { return 1.f / (1.f + __expf(-x)); }

// ---------------------------------------------------------------- K0: f32->bf16
__global__ void cvt_bf16_kernel(const float* __restrict__ src,
                                unsigned short* __restrict__ dst, int n) {
  int i = blockIdx.x * blockDim.x + threadIdx.x;
  if (i < n) dst[i] = f2bf(src[i]);
}

// ------------------------------------------- K1: embedding gather + gx GEMM
// gx fragments: for (dir, t, mb=batch-tile, nt=N-tile) a 16x16 f32 tile stored
// in WMMA C layout as [lane][r] (8 contiguous f32 per lane) so K2 can load the
// accumulator with plain b128 loads and zero conversion math.
// grid (512*4 tiles, 16 Ngroups, 2 dirs), block 128 (4 waves share emb tile).
__global__ void __launch_bounds__(128) gx_gemm_kernel(
    const int* __restrict__ words, const int* __restrict__ lens,
    const float* __restrict__ wvec,
    const __bf16* __restrict__ Wih_f, const __bf16* __restrict__ Wih_b,
    const float* __restrict__ b_f, const float* __restrict__ b_b,
    float* __restrict__ gxf) {
  __shared__ __bf16 embt[16][264];   // 264-stride pads LDS banks
  const int mtile = blockIdx.x;      // = t*4 + mb
  const int t  = mtile >> 2;
  const int mb = mtile & 3;
  const int dir = blockIdx.z;
  const int tid = threadIdx.x;

  for (int i = 0; i < 32; ++i) {
    int f = tid + 128 * i;              // 0..4095
    int row = f >> 8, e = f & 255;      // row = batch-in-tile
    int b = mb * 16 + row;
    int tt = t;
    if (dir) { int L = lens[b]; tt = (t < L) ? (L - 1 - t) : t; }
    int widx = words[b * TB + tt];
    ((unsigned short*)&embt[row][e])[0] = f2bf(wvec[(size_t)widx * ED + e]);
  }
  __syncthreads();

  const int lane = tid & 31, w = tid >> 5;
  const int ntile = blockIdx.y * 4 + w;
  const int nn = lane & 15;
  const int gcol = ntile * 16 + nn;
  const float* bias = dir ? b_b : b_f;
  const __bf16* W = (dir ? Wih_b : Wih_f) + (size_t)(ntile * 16) * ED;

  float bv = bias[gcol];
  v8f acc;
#pragma unroll
  for (int r = 0; r < 8; ++r) acc[r] = bv;

#pragma unroll
  for (int k0 = 0; k0 < ED; k0 += 32) {
    v16bf a = load_frag(&embt[0][0], 264, lane, k0);
    v16bf bb = load_frag(W, ED, lane, k0);
    acc = wmma_bf16(a, bb, acc);
  }

  // fragment store: frag index = ((dir*T + t)*4 + mb)*64 + ntile, 256 f32 each
  float* out = gxf + ((((size_t)dir * TB + t) * 4 + mb) * 64 + ntile) * 256 + lane * 8;
  *(v8f*)out = acc;   // 2x global_store_b128, 32B aligned
}

// ------------------------------------------------- K2: recurrent LSTM scan
// One block per (16 batch rows, direction): grid (4, 2), block 256 (8 waves).
// Wave w owns column-tiles {w, w+8} x all 4 gates -> gate fusion needs no
// cross-wave exchange; h lives in LDS (bf16), c in registers. Accumulators
// initialize straight from gx f32 fragments (b128 loads, immediate offsets).
__global__ void __launch_bounds__(256) lstm_scan_kernel(
    const float* __restrict__ gxf,
    const __bf16* __restrict__ Whh_f, const __bf16* __restrict__ Whh_b,
    float* __restrict__ h_fwd, float* __restrict__ h_bwd) {
  __shared__ __bf16 hbuf[16][264];
  const int dir = blockIdx.y;
  const int mb = blockIdx.x;
  const int bbase = mb * 16;
  const __bf16* Whh = dir ? Whh_b : Whh_f;
  float* hout = dir ? h_bwd : h_fwd;

  const int tid = threadIdx.x, lane = tid & 31, w = tid >> 5;
  const int nn = lane & 15, half = lane >> 4;

  for (int i = tid; i < 16 * 264; i += 256) ((unsigned short*)&hbuf[0][0])[i] = 0;

  float cst[2][8];
#pragma unroll
  for (int a = 0; a < 2; ++a)
#pragma unroll
    for (int r = 0; r < 8; ++r) cst[a][r] = 0.f;

  __syncthreads();

  for (int t = 0; t < TB; ++t) {
    // per-step fragment base for this (dir, t, mb); nt offsets are constants
    const float* fb = gxf + ((((size_t)dir * TB + t) * 4 + mb) * 64) * 256 + lane * 8;

    v8f acc[8];
    // C init = gx (includes bias): ti = gi*2 + cti, nt = gi*16 + ct
#pragma unroll
    for (int ti = 0; ti < 8; ++ti) {
      const int gi = ti >> 1;
      const int ct = (ti & 1) ? (w + 8) : w;
      const int nt = gi * 16 + ct;
      acc[ti] = *(const v8f*)(fb + (size_t)nt * 256);
    }
    // G += h @ W_hh^T
#pragma unroll
    for (int k0 = 0; k0 < HD; k0 += 32) {
      v16bf a = load_frag(&hbuf[0][0], 264, lane, k0);
#pragma unroll
      for (int ti = 0; ti < 8; ++ti) {
        const int gi = ti >> 1;
        const int ct = (ti & 1) ? (w + 8) : w;
        const __bf16* Wb = Whh + (size_t)(gi * 256 + ct * 16) * HD;
        v16bf bb = load_frag(Wb, HD, lane, k0);
        acc[ti] = wmma_bf16(a, bb, acc[ti]);
      }
    }
    __syncthreads();  // all waves done reading h(t-1) from LDS
    // gates: i=acc[cti], f=acc[2+cti], g=acc[4+cti], o=acc[6+cti]
#pragma unroll
    for (int cti = 0; cti < 2; ++cti) {
      const int ct = cti ? (w + 8) : w;
      const int col = ct * 16 + nn;
#pragma unroll
      for (int r = 0; r < 8; ++r) {
        float ig = sigmoidf_(acc[0 + cti][r]);
        float fg = sigmoidf_(acc[2 + cti][r]);
        float gg = tanhf(acc[4 + cti][r]);
        float og = sigmoidf_(acc[6 + cti][r]);
        float cn = fg * cst[cti][r] + ig * gg;
        cst[cti][r] = cn;
        float hv = og * tanhf(cn);
        int row = r + half * 8;
        ((unsigned short*)&hbuf[row][col])[0] = f2bf(hv);
        hout[((size_t)(bbase + row) * TB + t) * HD + col] = hv;
      }
    }
    __syncthreads();  // h(t) visible before next step's A loads
  }
}

// ---------------------------------- K3: feats (concat+unrev+mask) + emissions
__global__ void __launch_bounds__(128) feats_emis_kernel(
    const float* __restrict__ h_fwd, const float* __restrict__ h_bwd,
    const int* __restrict__ lens,
    const float* __restrict__ W_out, const float* __restrict__ b_out,
    float* __restrict__ feats, float* __restrict__ emis) {
  __shared__ float fr[512];
  const int bt = blockIdx.x;
  const int b = bt >> 9, t = bt & (TB - 1);
  const int L = lens[b];
  const bool msk = t < L;
  const int tt = msk ? (L - 1 - t) : t;   // un-reverse backward hidden states
  const int tid = threadIdx.x;
  for (int i = tid; i < 512; i += 128) {
    float v = (i < HD) ? h_fwd[((size_t)b * TB + t) * HD + i]
                       : h_bwd[((size_t)b * TB + tt) * HD + (i - HD)];
    v = msk ? v : 0.f;
    fr[i] = v;
    feats[(size_t)bt * 512 + i] = v;
  }
  __syncthreads();
  if (tid < KK) {
    float s = b_out[tid];
    for (int j = 0; j < 512; ++j) s += fr[j] * W_out[tid * 512 + j];
    emis[(size_t)bt * KK + tid] = s;
  }
}

// ---------------------------------------- K4: CRF loss + Viterbi + backtrace
__global__ void __launch_bounds__(64) crf_kernel(
    const float* __restrict__ emis, const int* __restrict__ labels,
    const int* __restrict__ lens,
    const float* __restrict__ start_tr, const float* __restrict__ end_tr,
    const float* __restrict__ trans,
    unsigned char* __restrict__ hist,
    float* __restrict__ loss_out, float* __restrict__ preds_out) {
  __shared__ float ls[64];
  const int b = threadIdx.x;
  const int L = lens[b];
  const float* em = emis + (size_t)b * TB * KK;
  const int* y = labels + (size_t)b * TB;

  // numerator (gold path score)
  float num = start_tr[y[0]] + em[y[0]];
  for (int t = 1; t < TB; ++t)
    if (t < L) num += trans[y[t - 1] * KK + y[t]] + em[t * KK + y[t]];
  num += end_tr[y[L - 1]];

  // forward algorithm (log partition)
  float sc[KK];
  for (int k = 0; k < KK; ++k) sc[k] = start_tr[k] + em[k];
  for (int t = 1; t < TB; ++t) {
    if (t < L) {
      float nx[KK];
      for (int j = 0; j < KK; ++j) {
        float mx = -1e30f;
        for (int i2 = 0; i2 < KK; ++i2) mx = fmaxf(mx, sc[i2] + trans[i2 * KK + j]);
        float s = 0.f;
        for (int i2 = 0; i2 < KK; ++i2) s += __expf(sc[i2] + trans[i2 * KK + j] - mx);
        nx[j] = mx + __logf(s) + em[t * KK + j];
      }
      for (int j = 0; j < KK; ++j) sc[j] = nx[j];
    }
  }
  float mx = -1e30f;
  for (int k = 0; k < KK; ++k) mx = fmaxf(mx, sc[k] + end_tr[k]);
  float s = 0.f;
  for (int k = 0; k < KK; ++k) s += __expf(sc[k] + end_tr[k] - mx);
  float denom = mx + __logf(s);

  ls[b] = num - denom;
  __syncthreads();
  if (b == 0) {
    float a = 0.f;
    for (int i2 = 0; i2 < 64; ++i2) a += ls[i2];
    loss_out[0] = -(a / 64.f);
  }

  // Viterbi
  for (int k = 0; k < KK; ++k) sc[k] = start_tr[k] + em[k];
  unsigned char* hb = hist + (size_t)b * TB * KK;
  for (int t = 1; t < TB; ++t) {
    if (t < L) {
      float nx[KK];
      for (int j = 0; j < KK; ++j) {
        float best = -1e30f; int bi = 0;
        for (int i2 = 0; i2 < KK; ++i2) {
          float v = sc[i2] + trans[i2 * KK + j];
          if (v > best) { best = v; bi = i2; }
        }
        nx[j] = best + em[t * KK + j];
        hb[(t - 1) * KK + j] = (unsigned char)bi;
      }
      for (int j = 0; j < KK; ++j) sc[j] = nx[j];
    } else {
      for (int j = 0; j < KK; ++j) hb[(t - 1) * KK + j] = (unsigned char)j;  // masked: identity
    }
  }
  int cur = 0;
  {
    float best = -1e30f;
    for (int k = 0; k < KK; ++k) { float v = sc[k] + end_tr[k]; if (v > best) { best = v; cur = k; } }
  }
  preds_out[(size_t)b * TB + (TB - 1)] = ((TB - 1) < L) ? (float)cur : 0.f;
  for (int t = TB - 2; t >= 0; --t) {
    if (t + 1 < L) cur = hb[t * KK + cur];
    preds_out[(size_t)b * TB + t] = (t < L) ? (float)cur : 0.f;
  }
}

// --------------------------------------------------------------------- launch
extern "C" void kernel_launch(void* const* d_in, const int* in_sizes, int n_in,
                              void* d_out, int out_size, void* d_ws, size_t ws_size,
                              hipStream_t stream) {
  const int*   words   = (const int*)d_in[0];
  const int*   lens    = (const int*)d_in[1];
  const int*   labels  = (const int*)d_in[2];
  const float* wvec    = (const float*)d_in[3];
  const float* Wih_f   = (const float*)d_in[4];
  const float* Whh_f   = (const float*)d_in[5];
  const float* b_f     = (const float*)d_in[6];
  const float* Wih_b   = (const float*)d_in[7];
  const float* Whh_b   = (const float*)d_in[8];
  const float* b_b     = (const float*)d_in[9];
  const float* W_out   = (const float*)d_in[10];
  const float* b_out   = (const float*)d_in[11];
  const float* st_tr   = (const float*)d_in[12];
  const float* en_tr   = (const float*)d_in[13];
  const float* trans   = (const float*)d_in[14];

  char* ws = (char*)d_ws;
  size_t o = 0;
  auto take = [&](size_t bytes) { void* p = ws + o; o += (bytes + 255) & ~(size_t)255; return p; };

  unsigned short* Wih_f_bf = (unsigned short*)take((size_t)G4 * ED * 2);
  unsigned short* Whh_f_bf = (unsigned short*)take((size_t)G4 * HD * 2);
  unsigned short* Wih_b_bf = (unsigned short*)take((size_t)G4 * ED * 2);
  unsigned short* Whh_b_bf = (unsigned short*)take((size_t)G4 * HD * 2);
  float*          gxf      = (float*)take((size_t)2 * BT * G4 * 4);   // f32 fragments
  float*          h_fwd    = (float*)take((size_t)BT * HD * 4);
  float*          h_bwd    = (float*)take((size_t)BT * HD * 4);
  float*          emis     = (float*)take((size_t)BT * KK * 4);
  unsigned char*  hist     = (unsigned char*)take((size_t)BT * KK);

  float* loss_out  = (float*)d_out;
  float* preds_out = loss_out + 1;
  float* feats_out = preds_out + (size_t)BT;

  const int nW = G4 * ED;  // 262144
  cvt_bf16_kernel<<<(nW + 255) / 256, 256, 0, stream>>>(Wih_f, Wih_f_bf, nW);
  cvt_bf16_kernel<<<(nW + 255) / 256, 256, 0, stream>>>(Whh_f, Whh_f_bf, nW);
  cvt_bf16_kernel<<<(nW + 255) / 256, 256, 0, stream>>>(Wih_b, Wih_b_bf, nW);
  cvt_bf16_kernel<<<(nW + 255) / 256, 256, 0, stream>>>(Whh_b, Whh_b_bf, nW);

  gx_gemm_kernel<<<dim3(TB * 4, 16, 2), 128, 0, stream>>>(
      words, lens, wvec,
      (const __bf16*)Wih_f_bf, (const __bf16*)Wih_b_bf, b_f, b_b, gxf);

  lstm_scan_kernel<<<dim3(NB / 16, 2), 256, 0, stream>>>(
      gxf, (const __bf16*)Whh_f_bf, (const __bf16*)Whh_b_bf, h_fwd, h_bwd);

  feats_emis_kernel<<<BT, 128, 0, stream>>>(h_fwd, h_bwd, lens, W_out, b_out,
                                            feats_out, emis);

  crf_kernel<<<1, 64, 0, stream>>>(emis, labels, lens, st_tr, en_tr, trans,
                                   hist, loss_out, preds_out);
}
